// GraphSYMKANEncoder_11072425689331
// MI455X (gfx1250) — matrix operand order repaired and checked by reading, not compiled, not run on hardware
//
#include <hip/hip_runtime.h>

// Problem dims (fixed by reference)
#define N_NODES 32768
#define N_EDGES 524288
#define FNODE   64
#define HDIM    128
#define FEDGE   32
#define LAYERS  3
#define DMODEL  256
#define K1      (2*HDIM + FEDGE)   // 288  (msg MLP input)
#define H2      (2*HDIM)           // 256  (msg MLP hidden)

typedef __attribute__((ext_vector_type(16))) __bf16 v16bf;
typedef __attribute__((ext_vector_type(8)))  __bf16 v8bf;
typedef __attribute__((ext_vector_type(8)))  float  v8f;

#if defined(__HIP_DEVICE_COMPILE__) && __has_builtin(__builtin_amdgcn_global_load_async_to_lds_b128)
#define HAVE_ASYNC_LDS 1
#else
#define HAVE_ASYNC_LDS 0
#endif

#if HAVE_ASYNC_LDS
// From the compiler diagnostic: param0 is 'int __vector_size__(16) __device__ *'
// (= v4i in addrspace(1)); LDS side is the addrspace(3) analog.
typedef int vi4 __attribute__((vector_size(16)));
typedef __attribute__((address_space(1))) vi4 gvi4;
typedef __attribute__((address_space(3))) vi4 lvi4;
static __device__ __forceinline__ void async_copy16(const void* gsrc, void* ldst) {
  __builtin_amdgcn_global_load_async_to_lds_b128(
      (gvi4*)(void*)const_cast<void*>(gsrc), (lvi4*)ldst, 0, 0);
}
static __device__ __forceinline__ void wait_async0() {
#if __has_builtin(__builtin_amdgcn_s_wait_asynccnt)
  __builtin_amdgcn_s_wait_asynccnt(0);
#else
  asm volatile("s_wait_asynccnt 0x0" ::: "memory");
#endif
}
#endif

// ---------------- WMMA fragment helpers (CDNA5 16x16x32 bf16) ----------------
// A (16xK) 16-bit layout: lane L (m=L&15, hh=L>>4) holds
//   data[0..7]  = A[m][K0 + 8*hh + i],  data[8..15] = A[m][K0 + 16 + 8*hh + i]
// B (Kx16) 16-bit layout: lane L holds data[0..7] = B[K0 + 8*hh + i][m], data[8..15] = B[K0 + 16 + 8*hh + i][m]
//   -> with B stored transposed (N-major), two contiguous 16B runs per lane.
// C (16x16) f32 layout: lane L holds c[r] = C[r + 8*hh][m]

static __device__ __forceinline__ v16bf mk16(v8bf lo, v8bf hi) {
  return __builtin_shufflevector(lo, hi, 0,1,2,3,4,5,6,7,8,9,10,11,12,13,14,15);
}
static __device__ __forceinline__ v16bf load_a_bf16(const __bf16* row, int k0, int hh) {
  v8bf lo = *(const v8bf*)(row + k0 + hh*8);
  v8bf hi = *(const v8bf*)(row + k0 + 16 + hh*8);
  return mk16(lo, hi);
}
static __device__ __forceinline__ v16bf load_a_f32(const float* row, int k0, int hh) {
  v16bf r;
  #pragma unroll
  for (int i = 0; i < 8; ++i) {
    r[i]   = (__bf16)row[k0 + hh*8 + i];
    r[8+i] = (__bf16)row[k0 + 16 + hh*8 + i];
  }
  return r;
}
static __device__ __forceinline__ v16bf load_b_bf16(const __bf16* rowT, int k0, int hh) {
  v8bf lo = *(const v8bf*)(rowT + k0 + hh*8);
  v8bf hi = *(const v8bf*)(rowT + k0 + 16 + hh*8);
  return mk16(lo, hi);
}
static __device__ __forceinline__ v8f wmma_bf16(v16bf a, v16bf b, v8f c) {
  return __builtin_amdgcn_wmma_f32_16x16x32_bf16(
      false, a, false, b, (short)0, c, false, false);
}

// ---------------- Prep kernels ----------------
__global__ void k_cvt_bf16(const float* __restrict__ src, __bf16* __restrict__ dst, int n) {
  int i = blockIdx.x * blockDim.x + threadIdx.x;
  if (i < n) dst[i] = (__bf16)src[i];
}
__global__ void k_transpose_cvt(const float* __restrict__ src, __bf16* __restrict__ dst,
                                int R, int C) {
  int i = blockIdx.x * blockDim.x + threadIdx.x;
  if (i >= R * C) return;
  int r = i / C, c = i % C;
  dst[c * R + r] = (__bf16)src[i];   // dst is [C][R] (out-feature major)
}
__global__ void k_zero_f32(float* __restrict__ p, int n) {
  int i = blockIdx.x * blockDim.x + threadIdx.x;
  if (i < n) p[i] = 0.f;
}
__global__ void k_fill_ones(float* __restrict__ p, int n) {
  int i = blockIdx.x * blockDim.x + threadIdx.x;
  if (i < n) p[i] = 1.0f;
}

// ---------------- Node embedding: h = x @ W + b ----------------
__global__ __launch_bounds__(256)
void k_node_embed(const __bf16* __restrict__ xb, const __bf16* __restrict__ WT,
                  const float* __restrict__ bias,
                  float* __restrict__ hf, __bf16* __restrict__ hb) {
  int w = threadIdx.x >> 5, lane = threadIdx.x & 31;
  int m = lane & 15, hh = lane >> 4;
  int tile = blockIdx.x * 8 + w;
  const __bf16* rowX = xb + (long)(tile*16 + m) * FNODE;
  v16bf afr[FNODE/32];
  #pragma unroll
  for (int kk = 0; kk < FNODE/32; ++kk) afr[kk] = load_a_bf16(rowX, kk*32, hh);
  #pragma unroll
  for (int t = 0; t < HDIM/16; ++t) {
    const __bf16* wrow = WT + (t*16 + m) * FNODE;
    v8f c = {};
    #pragma unroll
    for (int kk = 0; kk < FNODE/32; ++kk)
      c = wmma_bf16(afr[kk], load_b_bf16(wrow, kk*32, hh), c);
    float bv = bias[t*16 + m];
    #pragma unroll
    for (int r = 0; r < 8; ++r) {
      int node = tile*16 + r + 8*hh;
      float v = c[r] + bv;
      hf[(long)node*HDIM + t*16 + m] = v;
      hb[(long)node*HDIM + t*16 + m] = (__bf16)v;
    }
  }
}

// ---------------- Fused edge MLP + scatter-add ----------------
// One wave = 16 edges. GEMM1: [16,288]@[288,256] -> relu -> LDS (bf16),
// GEMM2: [16,256]@[256,128] (B staged in LDS via async DMA) -> atomicAdd into aggr[dst].
__global__ __launch_bounds__(256)
void k_edge_mlp(const __bf16* __restrict__ hb, const __bf16* __restrict__ eb,
                const __bf16* __restrict__ W1T, const float* __restrict__ bias1,
                const __bf16* __restrict__ W2T, const float* __restrict__ bias2,
                const int* __restrict__ srcp, const int* __restrict__ dstp,
                float* __restrict__ aggr) {
  __shared__ __align__(16) __bf16 zl[8][16][H2];     // 64 KB: relu(z) transpose buffer
  __shared__ __align__(16) __bf16 w2l[HDIM][H2];     // 64 KB: W2T staged per workgroup
  int w = threadIdx.x >> 5, lane = threadIdx.x & 31;
  int m = lane & 15, hh = lane >> 4;
  int ebase = (blockIdx.x * 8 + w) * 16;

  // Kick off async DMA of this layer's W2T (64KB) into LDS; overlaps with GEMM1.
  {
    const int CH = (HDIM * H2 * (int)sizeof(__bf16)) / 16;   // 4096 x 16B chunks
#if HAVE_ASYNC_LDS
    const char* gsrc = (const char*)W2T;
    char* ldst = (char*)&w2l[0][0];
    for (int i = threadIdx.x; i < CH; i += 256)
      async_copy16(gsrc + (long)i*16, ldst + (long)i*16);
#else
    v8bf* ldst = (v8bf*)&w2l[0][0];
    const v8bf* gsrc = (const v8bf*)W2T;
    for (int i = threadIdx.x; i < CH; i += 256) ldst[i] = gsrc[i];
#endif
  }

  int di = dstp[ebase + m];
  int si = srcp[ebase + m];
  const __bf16* rowD = hb + (long)di * HDIM;
  const __bf16* rowS = hb + (long)si * HDIM;
  const __bf16* rowE = eb + (long)(ebase + m) * FEDGE;

  int nd[8];
  #pragma unroll
  for (int r = 0; r < 8; ++r) nd[r] = dstp[ebase + r + 8*hh];

  // Gather A fragments once (16 edges x 288 features -> 9 fragments in registers)
  v16bf afr[K1/32];
  #pragma unroll
  for (int kk = 0; kk < 4; ++kk) afr[kk]   = load_a_bf16(rowD, kk*32, hh);
  #pragma unroll
  for (int kk = 0; kk < 4; ++kk) afr[4+kk] = load_a_bf16(rowS, kk*32, hh);
  afr[8] = load_a_bf16(rowE, 0, hh);

  // GEMM1 over 16 output tiles (256 hidden features)
  for (int t = 0; t < H2/16; ++t) {
    const __bf16* wrow = W1T + (t*16 + m) * K1;
    __builtin_prefetch(wrow + 16*K1, 0, 1);
    v8f c = {};
    #pragma unroll
    for (int kk = 0; kk < K1/32; ++kk)
      c = wmma_bf16(afr[kk], load_b_bf16(wrow, kk*32, hh), c);
    float bv = bias1[t*16 + m];
    #pragma unroll
    for (int r = 0; r < 8; ++r)
      zl[w][r + 8*hh][t*16 + m] = (__bf16)fmaxf(c[r] + bv, 0.f);
  }

#if HAVE_ASYNC_LDS
  wait_async0();            // W2 DMA complete (this wave's requests)
#endif
  __syncthreads();          // zl transpose hand-off + cross-wave W2 visibility

  // GEMM2: A from zl (LDS), B from w2l (LDS)
  v16bf zfr[H2/32];
  const __bf16* zrow = &zl[w][m][0];
  #pragma unroll
  for (int kk = 0; kk < H2/32; ++kk) zfr[kk] = load_a_bf16(zrow, kk*32, hh);

  for (int t = 0; t < HDIM/16; ++t) {
    const __bf16* wrow = &w2l[t*16 + m][0];
    v8f c = {};
    #pragma unroll
    for (int kk = 0; kk < H2/32; ++kk)
      c = wmma_bf16(zfr[kk], load_b_bf16(wrow, kk*32, hh), c);
    float bv = bias2[t*16 + m];
    #pragma unroll
    for (int r = 0; r < 8; ++r)
      atomicAdd(&aggr[(long)nd[r]*HDIM + t*16 + m], c[r] + bv);
  }
}

// ---------------- Update: u = [h|aggr] @ Wu + b; ReLU; LN; residual; LN ----------------
__global__ __launch_bounds__(256)
void k_update(float* __restrict__ hf, __bf16* __restrict__ hb,
              const float* __restrict__ aggr,
              const __bf16* __restrict__ WuT, const float* __restrict__ ub,
              const float* __restrict__ g1, const float* __restrict__ bb1,
              const float* __restrict__ g2, const float* __restrict__ bb2) {
  __shared__ __align__(16) float ul[8][16][HDIM];   // 64 KB
  int w = threadIdx.x >> 5, lane = threadIdx.x & 31;
  int m = lane & 15, hh = lane >> 4;
  int tile = blockIdx.x * 8 + w;
  const __bf16* rowH = hb   + (long)(tile*16 + m) * HDIM;
  const float*  rowA = aggr + (long)(tile*16 + m) * HDIM;

  v16bf afr[H2/32];
  #pragma unroll
  for (int kk = 0; kk < 4; ++kk) afr[kk]   = load_a_bf16(rowH, kk*32, hh);
  #pragma unroll
  for (int kk = 0; kk < 4; ++kk) afr[4+kk] = load_a_f32(rowA, kk*32, hh);

  #pragma unroll
  for (int t = 0; t < HDIM/16; ++t) {
    const __bf16* wrow = WuT + (t*16 + m) * H2;
    v8f c = {};
    #pragma unroll
    for (int kk = 0; kk < H2/32; ++kk)
      c = wmma_bf16(afr[kk], load_b_bf16(wrow, kk*32, hh), c);
    float bv = ub[t*16 + m];
    #pragma unroll
    for (int r = 0; r < 8; ++r)
      ul[w][r + 8*hh][t*16 + m] = c[r] + bv;
  }
  __syncthreads();

  if (lane < 16) {
    int node = tile*16 + lane;
    float* ur = ul[w][lane];
    float mu = 0.f;
    for (int f = 0; f < HDIM; ++f) mu += fmaxf(ur[f], 0.f);
    mu *= (1.f/HDIM);
    float var = 0.f;
    for (int f = 0; f < HDIM; ++f) { float v = fmaxf(ur[f], 0.f) - mu; var += v*v; }
    var *= (1.f/HDIM);
    float rs = rsqrtf(var + 1e-5f);
    float mu2 = 0.f;
    for (int f = 0; f < HDIM; ++f) {
      float tv = hf[(long)node*HDIM + f] + (fmaxf(ur[f], 0.f) - mu) * rs * g1[f] + bb1[f];
      ur[f] = tv; mu2 += tv;
    }
    mu2 *= (1.f/HDIM);
    float var2 = 0.f;
    for (int f = 0; f < HDIM; ++f) { float v = ur[f] - mu2; var2 += v*v; }
    var2 *= (1.f/HDIM);
    float rs2 = rsqrtf(var2 + 1e-5f);
    for (int f = 0; f < HDIM; ++f) {
      float o = (ur[f] - mu2) * rs2 * g2[f] + bb2[f];
      hf[(long)node*HDIM + f] = o;
      hb[(long)node*HDIM + f] = (__bf16)o;
    }
  }
}

// ---------------- Projection: out = h @ Wp + b (dense batch is identity reshape) ----------------
__global__ __launch_bounds__(256)
void k_proj(const __bf16* __restrict__ hb, const __bf16* __restrict__ WpT,
            const float* __restrict__ bias, float* __restrict__ out) {
  int w = threadIdx.x >> 5, lane = threadIdx.x & 31;
  int m = lane & 15, hh = lane >> 4;
  int tile = blockIdx.x * 8 + w;
  const __bf16* rowH = hb + (long)(tile*16 + m) * HDIM;
  v16bf afr[HDIM/32];
  #pragma unroll
  for (int kk = 0; kk < HDIM/32; ++kk) afr[kk] = load_a_bf16(rowH, kk*32, hh);
  #pragma unroll
  for (int t = 0; t < DMODEL/16; ++t) {
    const __bf16* wrow = WpT + (t*16 + m) * HDIM;
    v8f c = {};
    #pragma unroll
    for (int kk = 0; kk < HDIM/32; ++kk)
      c = wmma_bf16(afr[kk], load_b_bf16(wrow, kk*32, hh), c);
    float bv = bias[t*16 + m];
    #pragma unroll
    for (int r = 0; r < 8; ++r) {
      int node = tile*16 + r + 8*hh;
      out[(long)node*DMODEL + t*16 + m] = c[r] + bv;
    }
  }
}

// ---------------- Host launcher ----------------
extern "C" void kernel_launch(void* const* d_in, const int* in_sizes, int n_in,
                              void* d_out, int out_size, void* d_ws, size_t ws_size,
                              hipStream_t stream) {
  const float* x        = (const float*)d_in[0];
  const float* eattr    = (const float*)d_in[1];
  const float* emb_w    = (const float*)d_in[2];
  const float* emb_b    = (const float*)d_in[3];
  const float* msg_w1   = (const float*)d_in[4];
  const float* msg_b1   = (const float*)d_in[5];
  const float* msg_w2   = (const float*)d_in[6];
  const float* msg_b2   = (const float*)d_in[7];
  const float* upd_w    = (const float*)d_in[8];
  const float* upd_b    = (const float*)d_in[9];
  const float* uln_g    = (const float*)d_in[10];
  const float* uln_b    = (const float*)d_in[11];
  const float* nrm_g    = (const float*)d_in[12];
  const float* nrm_b    = (const float*)d_in[13];
  const float* proj_w   = (const float*)d_in[14];
  const float* proj_b   = (const float*)d_in[15];
  const int*   eidx     = (const int*)d_in[16];
  const int*   srcp     = eidx;
  const int*   dstp     = eidx + N_EDGES;
  float* out = (float*)d_out;

  // Workspace carve-up
  char* ws = (char*)d_ws;
  float*  h_f32 = (float*)ws;  ws += (size_t)N_NODES*HDIM*4;
  float*  aggr  = (float*)ws;  ws += (size_t)N_NODES*HDIM*4;
  __bf16* h_bf  = (__bf16*)ws; ws += (size_t)N_NODES*HDIM*2;
  __bf16* x_bf  = (__bf16*)ws; ws += (size_t)N_NODES*FNODE*2;
  __bf16* ea_bf = (__bf16*)ws; ws += (size_t)N_EDGES*FEDGE*2;
  __bf16* WembT = (__bf16*)ws; ws += (size_t)HDIM*FNODE*2;
  __bf16* W1T   = (__bf16*)ws; ws += (size_t)LAYERS*H2*K1*2;
  __bf16* W2T   = (__bf16*)ws; ws += (size_t)LAYERS*HDIM*H2*2;
  __bf16* WuT   = (__bf16*)ws; ws += (size_t)LAYERS*HDIM*H2*2;
  __bf16* WpT   = (__bf16*)ws; ws += (size_t)DMODEL*HDIM*2;

  const int TB = 256;
  auto blk = [](long n, int tb){ return (int)((n + tb - 1) / tb); };

  // --- prep: bf16 conversions + weight transposes ---
  k_cvt_bf16<<<blk((long)N_NODES*FNODE, TB), TB, 0, stream>>>(x, x_bf, N_NODES*FNODE);
  k_cvt_bf16<<<blk((long)N_EDGES*FEDGE, TB), TB, 0, stream>>>(eattr, ea_bf, N_EDGES*FEDGE);
  k_transpose_cvt<<<blk(FNODE*HDIM, TB), TB, 0, stream>>>(emb_w, WembT, FNODE, HDIM);
  for (int l = 0; l < LAYERS; ++l) {
    k_transpose_cvt<<<blk(K1*H2, TB), TB, 0, stream>>>(msg_w1 + (long)l*K1*H2, W1T + (long)l*H2*K1, K1, H2);
    k_transpose_cvt<<<blk(H2*HDIM, TB), TB, 0, stream>>>(msg_w2 + (long)l*H2*HDIM, W2T + (long)l*HDIM*H2, H2, HDIM);
    k_transpose_cvt<<<blk(H2*HDIM, TB), TB, 0, stream>>>(upd_w + (long)l*H2*HDIM, WuT + (long)l*HDIM*H2, H2, HDIM);
  }
  k_transpose_cvt<<<blk(HDIM*DMODEL, TB), TB, 0, stream>>>(proj_w, WpT, HDIM, DMODEL);

  // --- node embedding ---
  k_node_embed<<<N_NODES/(16*8), TB, 0, stream>>>(x_bf, WembT, emb_b, h_f32, h_bf);

  // --- message-passing layers ---
  for (int l = 0; l < LAYERS; ++l) {
    k_zero_f32<<<blk((long)N_NODES*HDIM, TB), TB, 0, stream>>>(aggr, N_NODES*HDIM);
    k_edge_mlp<<<N_EDGES/(16*8), TB, 0, stream>>>(
        h_bf, ea_bf,
        W1T + (long)l*H2*K1,  msg_b1 + (long)l*H2,
        W2T + (long)l*HDIM*H2, msg_b2 + (long)l*HDIM,
        srcp, dstp, aggr);
    k_update<<<N_NODES/(16*8), TB, 0, stream>>>(
        h_f32, h_bf, aggr,
        WuT + (long)l*HDIM*H2, upd_b + (long)l*HDIM,
        uln_g + (long)l*HDIM, uln_b + (long)l*HDIM,
        nrm_g + (long)l*HDIM, nrm_b + (long)l*HDIM);
  }

  // --- projection + mask ---
  k_proj<<<N_NODES/(16*8), TB, 0, stream>>>(h_bf, WpT, proj_b, out);
  long mcount = (long)out_size - (long)N_NODES*DMODEL;
  if (mcount > 0)
    k_fill_ones<<<blk(mcount, TB), TB, 0, stream>>>(out + (long)N_NODES*DMODEL, (int)mcount);
}
